// CrossAttentionLayer_58772332478608
// MI455X (gfx1250) — matrix operand order repaired and checked by reading, not compiled, not run on hardware
//
#include <hip/hip_runtime.h>
#include <hip/hip_bf16.h>
#include <math.h>

// ---------------------------------------------------------------------------
// CrossAttentionLayer for MI455X (gfx1250), bf16 WMMA + f32 accumulate.
// B=8, Q=K=1024, D=1024, H=16, HD=64, FFN=4096.
// ---------------------------------------------------------------------------

typedef __bf16 bf16;
typedef __attribute__((ext_vector_type(16))) __bf16 v16bf;
typedef __attribute__((ext_vector_type(8)))  float  v8f;

static constexpr int Bc = 8, Qc = 1024, Kc = 1024, Dc = 1024, Hc = 16, HDc = 64, FFNc = 4096;
static constexpr int Mrows = Bc * Qc;              // 8192

union FragU { v16bf v; uint4 u[2]; };

static __device__ __forceinline__ v8f wmma_bf16(const v16bf& a, const v16bf& b, const v8f& c) {
    return __builtin_amdgcn_wmma_f32_16x16x32_bf16(false, a, false, b, (short)0, c, false, false);
}

// -------------------------- fp32 -> bf16 cast (x4 vectorized) --------------
__global__ void cast_f32_bf16_kernel(const float4* __restrict__ src, uint2* __restrict__ dst, int n4) {
    int i = blockIdx.x * blockDim.x + threadIdx.x;
    if (i < n4) {
        float4 v = src[i];
        union { uint2 u; bf16 h[4]; } pk;
        pk.h[0] = (bf16)v.x; pk.h[1] = (bf16)v.y; pk.h[2] = (bf16)v.z; pk.h[3] = (bf16)v.w;
        dst[i] = pk.u;
    }
}

// -------------------------- LayerNorm over D=1024 --------------------------
__global__ void __launch_bounds__(256)
layernorm_kernel(const float* __restrict__ x, const float* __restrict__ w,
                 const float* __restrict__ bsh, bf16* __restrict__ out) {
    __shared__ float sred[2][8];
    const int row  = blockIdx.x;
    const int tid  = threadIdx.x;
    const int wv   = tid >> 5, ln = tid & 31;
    const float4 xv = reinterpret_cast<const float4*>(x + (size_t)row * Dc)[tid];
    float s  = xv.x + xv.y + xv.z + xv.w;
    float ss = xv.x*xv.x + xv.y*xv.y + xv.z*xv.z + xv.w*xv.w;
    #pragma unroll
    for (int off = 16; off; off >>= 1) {
        s  += __shfl_xor(s,  off, 32);
        ss += __shfl_xor(ss, off, 32);
    }
    if (ln == 0) { sred[0][wv] = s; sred[1][wv] = ss; }
    __syncthreads();
    float tot = 0.f, tot2 = 0.f;
    #pragma unroll
    for (int i = 0; i < 8; ++i) { tot += sred[0][i]; tot2 += sred[1][i]; }
    const float mu  = tot  * (1.f / Dc);
    const float var = tot2 * (1.f / Dc) - mu * mu;
    const float inv = rsqrtf(var + 1e-5f);
    const int c = tid * 4;
    const float xe[4] = { xv.x, xv.y, xv.z, xv.w };
    union { uint2 u; bf16 h[4]; } pk;
    #pragma unroll
    for (int j = 0; j < 4; ++j)
        pk.h[j] = (bf16)((xe[j] - mu) * inv * w[c + j] + bsh[c + j]);
    *reinterpret_cast<uint2*>(out + (size_t)row * Dc + c) = pk.u;
}

// -------------------------- WMMA GEMM: C = act(A*W^T + b) (+res) -----------
// A: (M,Kd) bf16 row-major.  W: (N,Kd) bf16 row-major.
// Block tile 64x256; wave tile 32x64 (2x4 WMMA accumulators).
// outB_trans: store bf16 output transposed as (row>>10, col, row&1023)  [V^T]
__global__ void __launch_bounds__(256)
gemm_bf16_kernel(const bf16* __restrict__ A, const bf16* __restrict__ W,
                 const float* __restrict__ bias,
                 float* __restrict__ outF, bf16* __restrict__ outB,
                 const float* __restrict__ residual,
                 int M, int N, int Kd, int act, int outB_trans) {
    const int tid  = threadIdx.x;
    const int wv   = tid >> 5;
    const int lane = tid & 31;
    const int lo   = lane & 15;
    const int hiA  = (lane & 16) ? 8  : 0;   // A K-group / C M-group offset
    const int hiB  = (lane & 16) ? 16 : 0;   // unused; B loads use hiA as K-group offset
    const int mg   = wv >> 2, ng = wv & 3;
    const int m0 = blockIdx.y * 64 + mg * 32;
    const int n0 = blockIdx.x * 256 + ng * 64;

    const bf16* Ar0 = A + (size_t)(m0 + lo) * Kd;
    const bf16* Ar1 = A + (size_t)(m0 + 16 + lo) * Kd;
    const bf16* Wr[4];
    #pragma unroll
    for (int t = 0; t < 4; ++t) Wr[t] = W + (size_t)(n0 + t * 16 + lo) * Kd;

    v8f acc[2][4] = {};
    for (int k0 = 0; k0 < Kd; k0 += 32) {
        FragU fa0, fa1, fb[4];
        fa0.u[0] = *reinterpret_cast<const uint4*>(Ar0 + k0 + hiA);
        fa0.u[1] = *reinterpret_cast<const uint4*>(Ar0 + k0 + 16 + hiA);
        fa1.u[0] = *reinterpret_cast<const uint4*>(Ar1 + k0 + hiA);
        fa1.u[1] = *reinterpret_cast<const uint4*>(Ar1 + k0 + 16 + hiA);
        #pragma unroll
        for (int t = 0; t < 4; ++t) {
            fb[t].u[0] = *reinterpret_cast<const uint4*>(Wr[t] + k0 + hiA);
            fb[t].u[1] = *reinterpret_cast<const uint4*>(Wr[t] + k0 + 16 + hiA);
        }
        #pragma unroll
        for (int t = 0; t < 4; ++t) {
            acc[0][t] = wmma_bf16(fa0.v, fb[t].v, acc[0][t]);
            acc[1][t] = wmma_bf16(fa1.v, fb[t].v, acc[1][t]);
        }
    }
    #pragma unroll
    for (int mt = 0; mt < 2; ++mt) {
        #pragma unroll
        for (int t = 0; t < 4; ++t) {
            const int col  = n0 + t * 16 + lo;
            const float bs = bias[col];
            const int row0 = m0 + mt * 16 + hiA;
            if (outB_trans) {       // fused V^T store: 8 consecutive K -> one b128
                union { uint4 u; bf16 h[8]; } st;
                #pragma unroll
                for (int r = 0; r < 8; ++r) st.h[r] = (bf16)(acc[mt][t][r] + bs);
                *reinterpret_cast<uint4*>(outB + ((size_t)(row0 >> 10) << 20) +
                                          ((size_t)col << 10) + (row0 & 1023)) = st.u;
            } else {
                #pragma unroll
                for (int r = 0; r < 8; ++r) {
                    const int row = row0 + r;
                    float v = acc[mt][t][r] + bs;
                    if (act) v = 0.5f * v * (1.0f + erff(v * 0.70710678118f)); // exact GELU
                    if (residual) v += residual[(size_t)row * N + col];
                    if (outF) outF[(size_t)row * N + col] = v;
                    if (outB) outB[(size_t)row * N + col] = (bf16)v;
                }
            }
        }
    }
}

// -------------------------- Attention: block per (b, 16-query tile) --------
// LDS: S  = 16x1024 f32 scores/exps (64KB)
//      Pb = 16x1024 bf16 normalized probs (32KB) -> b128 ds loads in P@V
__global__ void __launch_bounds__(256)
attention_kernel(const bf16* __restrict__ qp, const bf16* __restrict__ kp,
                 const bf16* __restrict__ vt, const unsigned char* __restrict__ mask,
                 bf16* __restrict__ attn_out, float* __restrict__ attnw) {
    extern __shared__ char smem[];
    float* S  = reinterpret_cast<float*>(smem);
    bf16*  Pb = reinterpret_cast<bf16*>(smem + 16 * Kc * sizeof(float));
    const int tid  = threadIdx.x;
    const int wv   = tid >> 5, lane = tid & 31, lo = lane & 15;
    const int hiA  = (lane & 16) ? 8  : 0;
    const int hiB  = (lane & 16) ? 16 : 0;
    const int b    = blockIdx.x >> 6;
    const int q0   = (blockIdx.x & 63) * 16;
    const float scale = 0.125f;                        // 1/sqrt(64)

    for (int h = 0; h < Hc; ++h) {
        // ---- phase 1: scores = scale * Q K^T, masked -> LDS (f32)
        const bf16* Qrow = qp + (((size_t)(b * Qc + q0 + lo)) << 10) + h * HDc;
        FragU fq0, fq1;
        fq0.u[0] = *(const uint4*)(Qrow + hiA);
        fq0.u[1] = *(const uint4*)(Qrow + 16 + hiA);
        fq1.u[0] = *(const uint4*)(Qrow + 32 + hiA);
        fq1.u[1] = *(const uint4*)(Qrow + 48 + hiA);
        for (int kt = wv; kt < 64; kt += 8) {
            const int key = kt * 16 + lo;
            const bf16* Krow = kp + (((size_t)(b * Kc + key)) << 10) + h * HDc;
            FragU fk0, fk1;
            fk0.u[0] = *(const uint4*)(Krow + hiA);
            fk0.u[1] = *(const uint4*)(Krow + 16 + hiA);
            fk1.u[0] = *(const uint4*)(Krow + 32 + hiA);
            fk1.u[1] = *(const uint4*)(Krow + 48 + hiA);
            v8f acc = {};
            acc = wmma_bf16(fq0.v, fk0.v, acc);
            acc = wmma_bf16(fq1.v, fk1.v, acc);
            const bool valid = mask[b * Kc + key] != 0;
            #pragma unroll
            for (int r = 0; r < 8; ++r)
                S[(hiA + r) * Kc + key] = valid ? acc[r] * scale : -__builtin_inff();
        }
        __syncthreads();
        // ---- phase 2: softmax (2 rows/wave, float4 passes), emit bf16 probs
        //      + head-averaged attention weights accumulated to global
        #pragma unroll
        for (int rr = 0; rr < 2; ++rr) {
            const int row = wv * 2 + rr;
            float* Sr = S + row * Kc;
            float mx = -__builtin_inff();
            #pragma unroll
            for (int i = 0; i < 8; ++i) {
                const float4 v4 = *reinterpret_cast<const float4*>(&Sr[lane * 4 + i * 128]);
                mx = fmaxf(mx, fmaxf(fmaxf(v4.x, v4.y), fmaxf(v4.z, v4.w)));
            }
            #pragma unroll
            for (int off = 16; off; off >>= 1) mx = fmaxf(mx, __shfl_xor(mx, off, 32));
            float sum = 0.f;
            #pragma unroll
            for (int i = 0; i < 8; ++i) {
                const int c = lane * 4 + i * 128;
                float4 v4 = *reinterpret_cast<const float4*>(&Sr[c]);
                v4.x = __expf(v4.x - mx); v4.y = __expf(v4.y - mx);
                v4.z = __expf(v4.z - mx); v4.w = __expf(v4.w - mx);
                sum += (v4.x + v4.y) + (v4.z + v4.w);
                *reinterpret_cast<float4*>(&Sr[c]) = v4;
            }
            #pragma unroll
            for (int off = 16; off; off >>= 1) sum += __shfl_xor(sum, off, 32);
            const float inv = 1.0f / sum;
            float* aw = attnw + (((size_t)(b * Qc + q0 + row)) << 10);
            bf16* Pr  = Pb + row * Kc;
            #pragma unroll
            for (int i = 0; i < 8; ++i) {
                const int c = lane * 4 + i * 128;
                const float4 e4 = *reinterpret_cast<const float4*>(&Sr[c]);
                float4 p4; p4.x = e4.x*inv; p4.y = e4.y*inv; p4.z = e4.z*inv; p4.w = e4.w*inv;
                union { uint2 u; bf16 h[4]; } pk;
                pk.h[0]=(bf16)p4.x; pk.h[1]=(bf16)p4.y; pk.h[2]=(bf16)p4.z; pk.h[3]=(bf16)p4.w;
                *reinterpret_cast<uint2*>(&Pr[c]) = pk.u;
                float4 acw;
                if (h == 0) { acw.x = acw.y = acw.z = acw.w = 0.f; }
                else acw = *reinterpret_cast<const float4*>(&aw[c]);   // same thread/addr per head
                acw.x += p4.x * (1.f/Hc); acw.y += p4.y * (1.f/Hc);
                acw.z += p4.z * (1.f/Hc); acw.w += p4.w * (1.f/Hc);
                *reinterpret_cast<float4*>(&aw[c]) = acw;
            }
        }
        __syncthreads();
        // ---- phase 3: out = P @ V  (wave -> 16-col tile x half of keys)
        const int ntile = wv & 3, khalf = wv >> 2;
        v8f acc = {};
        const bf16* Prow = Pb + lo * Kc;
        for (int kb = 0; kb < 512; kb += 32) {
            const int kk = khalf * 512 + kb;
            FragU fp, fv;
            fp.u[0] = *(const uint4*)(Prow + kk + hiA);         // ds_load_b128
            fp.u[1] = *(const uint4*)(Prow + kk + 16 + hiA);
            const bf16* Vr = vt + ((size_t)b << 20) + (size_t)(h * HDc + ntile * 16 + lo) * Kc + kk + hiA;
            fv.u[0] = *(const uint4*)(Vr);
            fv.u[1] = *(const uint4*)(Vr + 16);
            acc = wmma_bf16(fp.v, fv.v, acc);
        }
        float* red = S;                                // S dead; reuse (disjoint from Pb)
        if (khalf == 1) {
            #pragma unroll
            for (int r = 0; r < 8; ++r) red[(ntile * 32 + lane) * 8 + r] = acc[r];
        }
        __syncthreads();
        if (khalf == 0) {
            #pragma unroll
            for (int r = 0; r < 8; ++r) {
                const float v2 = acc[r] + red[(ntile * 32 + lane) * 8 + r];
                const int m = hiA + r;
                attn_out[(((size_t)(b * Qc + q0 + m)) << 10) + h * HDc + ntile * 16 + lo] = (bf16)v2;
            }
        }
        __syncthreads();
    }
}

// ---------------------------------------------------------------------------
extern "C" void kernel_launch(void* const* d_in, const int* in_sizes, int n_in,
                              void* d_out, int out_size, void* d_ws, size_t ws_size,
                              hipStream_t stream) {
    const float* query      = (const float*)d_in[0];
    const float* key_value  = (const float*)d_in[1];
    const unsigned char* kpm = (const unsigned char*)d_in[2];
    const float* ln_q_w  = (const float*)d_in[3];
    const float* ln_q_b  = (const float*)d_in[4];
    const float* ln_kv_w = (const float*)d_in[5];
    const float* ln_kv_b = (const float*)d_in[6];
    const float* ln_f_w  = (const float*)d_in[7];
    const float* ln_f_b  = (const float*)d_in[8];
    const float* in_proj_w  = (const float*)d_in[9];
    const float* in_proj_b  = (const float*)d_in[10];
    const float* out_proj_w = (const float*)d_in[11];
    const float* out_proj_b = (const float*)d_in[12];
    const float* ffn_w1 = (const float*)d_in[13];
    const float* ffn_b1 = (const float*)d_in[14];
    const float* ffn_w2 = (const float*)d_in[15];
    const float* ffn_b2 = (const float*)d_in[16];

    char* ws = (char*)d_ws;
    size_t off = 0;
    auto alloc = [&](size_t bytes) -> void* {
        void* p = ws + off; off += (bytes + 255) & ~(size_t)255; return p;
    };
    bf16* w_inproj = (bf16*)alloc((size_t)3 * Dc * Dc * sizeof(bf16));
    bf16* w_out    = (bf16*)alloc((size_t)Dc * Dc * sizeof(bf16));
    bf16* w_f1     = (bf16*)alloc((size_t)FFNc * Dc * sizeof(bf16));
    bf16* w_f2     = (bf16*)alloc((size_t)Dc * FFNc * sizeof(bf16));
    bf16* qnorm    = (bf16*)alloc((size_t)Mrows * Dc * sizeof(bf16));
    bf16* kvnorm   = (bf16*)alloc((size_t)Bc * Kc * Dc * sizeof(bf16));
    bf16* qp       = (bf16*)alloc((size_t)Mrows * Dc * sizeof(bf16));
    bf16* kp       = (bf16*)alloc((size_t)Bc * Kc * Dc * sizeof(bf16));
    bf16* vt       = (bf16*)alloc((size_t)Bc * Dc * Kc * sizeof(bf16));   // V^T (B,D,K)
    bf16* attn_o   = (bf16*)alloc((size_t)Mrows * Dc * sizeof(bf16));
    float* xres    = (float*)alloc((size_t)Mrows * Dc * sizeof(float));
    bf16* hnorm    = (bf16*)alloc((size_t)Mrows * Dc * sizeof(bf16));
    bf16* hbuf     = (bf16*)alloc((size_t)Mrows * FFNc * sizeof(bf16));

    float* out_x  = (float*)d_out;
    float* out_aw = (float*)d_out + (size_t)Mrows * Dc;   // (B,Q,K) section

    // 1) weights -> bf16
    {
        int n4;
        n4 = 3 * Dc * Dc / 4;  cast_f32_bf16_kernel<<<n4 / 256, 256, 0, stream>>>((const float4*)in_proj_w,  (uint2*)w_inproj, n4);
        n4 = Dc * Dc / 4;      cast_f32_bf16_kernel<<<n4 / 256, 256, 0, stream>>>((const float4*)out_proj_w, (uint2*)w_out,    n4);
        n4 = FFNc * Dc / 4;    cast_f32_bf16_kernel<<<n4 / 256, 256, 0, stream>>>((const float4*)ffn_w1,     (uint2*)w_f1,     n4);
        n4 = Dc * FFNc / 4;    cast_f32_bf16_kernel<<<n4 / 256, 256, 0, stream>>>((const float4*)ffn_w2,     (uint2*)w_f2,     n4);
    }
    // 2) LayerNorms on inputs
    layernorm_kernel<<<Mrows, 256, 0, stream>>>(query,     ln_q_w,  ln_q_b,  qnorm);
    layernorm_kernel<<<Bc * Kc, 256, 0, stream>>>(key_value, ln_kv_w, ln_kv_b, kvnorm);
    // 3) Q/K/V projections (V written directly transposed -> vt)
    {
        dim3 grid(Dc / 256, Mrows / 64);
        gemm_bf16_kernel<<<grid, 256, 0, stream>>>(qnorm,  w_inproj,                       in_proj_b,
                                                   nullptr, qp, nullptr, Mrows, Dc, Dc, 0, 0);
        gemm_bf16_kernel<<<grid, 256, 0, stream>>>(kvnorm, w_inproj + (size_t)Dc * Dc,     in_proj_b + Dc,
                                                   nullptr, kp, nullptr, Bc * Kc, Dc, Dc, 0, 0);
        gemm_bf16_kernel<<<grid, 256, 0, stream>>>(kvnorm, w_inproj + (size_t)2 * Dc * Dc, in_proj_b + 2 * Dc,
                                                   nullptr, vt, nullptr, Bc * Kc, Dc, Dc, 0, 1);
    }
    // 4) attention (scores + softmax + P@V), 96KB dynamic LDS
    attention_kernel<<<Bc * (Qc / 16), 256,
                       16 * Kc * (sizeof(float) + sizeof(bf16)), stream>>>(
        qp, kp, vt, kpm, attn_o, out_aw);
    // 5) out-proj + residual(query) -> x
    {
        dim3 grid(Dc / 256, Mrows / 64);
        gemm_bf16_kernel<<<grid, 256, 0, stream>>>(attn_o, w_out, out_proj_b,
                                                   xres, nullptr, query, Mrows, Dc, Dc, 0, 0);
    }
    // 6) final LN
    layernorm_kernel<<<Mrows, 256, 0, stream>>>(xres, ln_f_w, ln_f_b, hnorm);
    // 7) FFN1 + GELU
    {
        dim3 grid(FFNc / 256, Mrows / 64);
        gemm_bf16_kernel<<<grid, 256, 0, stream>>>(hnorm, w_f1, ffn_b1,
                                                   nullptr, hbuf, nullptr, Mrows, FFNc, Dc, 1, 0);
    }
    // 8) FFN2 + residual(x) -> d_out
    {
        dim3 grid(Dc / 256, Mrows / 64);
        gemm_bf16_kernel<<<grid, 256, 0, stream>>>(hbuf, w_f2, ffn_b2,
                                                   out_x, nullptr, xres, Mrows, Dc, FFNc, 0, 0);
    }
    (void)in_sizes; (void)n_in; (void)out_size; (void)ws_size;
}
